// BatchedGraphSAGEDynamicRangeMean1_40862318854443
// MI455X (gfx1250) — matrix operand order repaired and checked by reading, not compiled, not run on hardware
//
#include <hip/hip_runtime.h>
#include <hip/hip_bf16.h>
#include <math.h>

typedef __attribute__((ext_vector_type(16))) _Float16 v16h;
typedef __attribute__((ext_vector_type(8)))  float    v8f;

#define NB   8
#define NN   4096
#define DD   256
#define DOUTC 256
#define CC   512      // 2*DOUT
#define PP   16
#define TTW  8
#define MM   256      // NN/PP
#define WW   272      // PP*(2*TTW+1)
#define NROWS (NB*NN)       // 32768
#define NRT   (NROWS/16)    // 2048 row tiles

union F16Frag { v16h v; uint4 q[2]; };

// A-matrix 16x32 f16 fragment from a row-major [16 x 256] f16 matrix, k-step ks.
// Per ISA 7.12.2: lane&15 = row M; lanes<16 hold K 0..7 & 16..23, lanes>=16 hold 8..15 & 24..31.
__device__ __forceinline__ v16h load_a_frag(const _Float16* base, int lane, int ks) {
  int row = lane & 15;
  int off = ks * 32 + ((lane >> 4) << 3);
  const _Float16* p = base + row * DD + off;
  F16Frag u;
  u.q[0] = *(const uint4*)(p);        // K = off .. off+7
  u.q[1] = *(const uint4*)(p + 16);   // K = off+16 .. off+23
  return u.v;
}

// B-matrix 32x16 f16 fragment. Element B[k][n] = mat[(colBase+n)*256 + k].
// lane&15 = column n; lanes<16 hold K 0..7 & 16..23, lanes>=16 hold K 8..15 & 24..31.
__device__ __forceinline__ v16h load_b_frag(const _Float16* mat, int colBase, int lane, int ks) {
  int cr = colBase + (lane & 15);
  int d0 = ks * 32 + ((lane >> 4) << 3);
  const _Float16* p = mat + cr * DD + d0;
  F16Frag u;
  u.q[0] = *(const uint4*)(p);
  u.q[1] = *(const uint4*)(p + 16);
  return u.v;
}

// ---------------- K0a: x f32 -> f16 ----------------
__global__ __launch_bounds__(256) void cvt_x_kernel(const float* __restrict__ x,
                                                    _Float16* __restrict__ xh) {
  size_t i = ((size_t)blockIdx.x * 256 + threadIdx.x) * 4;
  float4 f = *(const float4*)(x + i);
  _Float16* d = xh + i;
  d[0] = (_Float16)f.x; d[1] = (_Float16)f.y; d[2] = (_Float16)f.z; d[3] = (_Float16)f.w;
}

// ---------------- K0b: weights f32 -> f16 ----------------
__global__ __launch_bounds__(256) void cvt_w_kernel(const float* __restrict__ wx,
                                                    const float* __restrict__ wn,
                                                    _Float16* __restrict__ wxh,
                                                    _Float16* __restrict__ wnh) {
  int i = blockIdx.x * 256 + threadIdx.x;   // 0..65535
  wxh[i] = (_Float16)wx[i];
  wnh[i] = (_Float16)wn[i];
}

// ---------------- K1: window normalize + WMMA sim + top-3 + neighbor mean ----------------
__global__ __launch_bounds__(256) void knn_neib_kernel(const float* __restrict__ x,
                                                       _Float16* __restrict__ nh) {
  extern __shared__ char smem[];
  _Float16* gn  = (_Float16*)smem;                                  // WW*DD f16 (139264 B)
  float*    sim = (float*)(smem + (size_t)WW * DD * sizeof(_Float16)); // 16*WW f32
  int*      topj = (int*)(sim + 16 * WW);                           // 16*2 ints

  const int m = blockIdx.x, b = blockIdx.y;
  const int tid = threadIdx.x, lane = tid & 31, wv = tid >> 5;
  const int base = (m - TTW) * PP;

  // ---- normalize 272 window rows into LDS f16 (one row per wave iteration) ----
  for (int r = wv; r < WW; r += 8) {
    int jj = base + r;
    int j = min(max(jj, 0), NN - 1);
    const float4* xr = (const float4*)(x + ((size_t)b * NN + j) * DD);
    float4 f0 = xr[lane * 2 + 0];
    float4 f1 = xr[lane * 2 + 1];
    float s = f0.x*f0.x + f0.y*f0.y + f0.z*f0.z + f0.w*f0.w
            + f1.x*f1.x + f1.y*f1.y + f1.z*f1.z + f1.w*f1.w;
    #pragma unroll
    for (int off = 16; off > 0; off >>= 1) s += __shfl_xor(s, off, 32);
    float invn = 1.f / fmaxf(sqrtf(s), 1e-12f);
    union { _Float16 h[8]; uint4 q; } tmp;
    tmp.h[0] = (_Float16)(f0.x * invn); tmp.h[1] = (_Float16)(f0.y * invn);
    tmp.h[2] = (_Float16)(f0.z * invn); tmp.h[3] = (_Float16)(f0.w * invn);
    tmp.h[4] = (_Float16)(f1.x * invn); tmp.h[5] = (_Float16)(f1.y * invn);
    tmp.h[6] = (_Float16)(f1.z * invn); tmp.h[7] = (_Float16)(f1.w * invn);
    *(uint4*)(gn + (size_t)r * DD + lane * 8) = tmp.q;
  }
  __syncthreads();

  // ---- sim[16 x 272] = center(16x256) x window^T via WMMA f16 ----
  const _Float16* gnc = gn + (size_t)(TTW * PP) * DD;   // center rows 128..143
  for (int i = 0; i < 3; ++i) {
    int tile = wv + 8 * i;             // uniform per wave
    if (tile >= 17) break;
    v8f acc = {};
    #pragma unroll
    for (int ks = 0; ks < 8; ++ks) {
      v16h a  = load_a_frag(gnc, lane, ks);
      v16h bf = load_b_frag(gn, tile * 16, lane, ks);
      acc = __builtin_amdgcn_wmma_f32_16x16x32_f16(false, a, false, bf,
                                                   (short)0, acc, false, false);
    }
    int wcol = tile * 16 + (lane & 15);
    int jj = base + wcol;
    bool valid = (jj >= 0) && (jj < NN);
    #pragma unroll
    for (int v = 0; v < 8; ++v) {
      int q = v + ((lane >> 4) << 3);
      sim[q * WW + wcol] = valid ? acc[v] : -__builtin_inff();
    }
  }
  __syncthreads();

  // ---- top-3 scan (lowest index wins ties, matching lax.top_k) ----
  if (tid < 16) {
    const float* srow = sim + tid * WW;
    float v0 = -__builtin_inff(), v1 = v0, v2 = v0;
    int i0 = 0, i1 = 0, i2 = 0;
    for (int w = 0; w < WW; ++w) {
      float s = srow[w];
      if (s > v0)      { v2 = v1; i2 = i1; v1 = v0; i1 = i0; v0 = s; i0 = w; }
      else if (s > v1) { v2 = v1; i2 = i1; v1 = s;  i1 = w; }
      else if (s > v2) { v2 = s;  i2 = w; }
    }
    topj[tid * 2 + 0] = min(max(base + i1, 0), NN - 1);
    topj[tid * 2 + 1] = min(max(base + i2, 0), NN - 1);
  }
  __syncthreads();

  // ---- x_neib = mean of neighbors 2 & 3, stored f16 ----
  {
    int q  = tid >> 4;
    int d0 = (tid & 15) * 16;
    int j1 = topj[q * 2 + 0];
    int j2 = topj[q * 2 + 1];
    const float* r1 = x + ((size_t)b * NN + j1) * DD + d0;
    const float* r2 = x + ((size_t)b * NN + j2) * DD + d0;
    _Float16* dst = nh + ((size_t)b * NN + m * PP + q) * DD + d0;
    #pragma unroll
    for (int k = 0; k < 16; ++k) dst[k] = (_Float16)(0.5f * (r1[k] + r2[k]));
  }
}

// ---------------- K2: dual GEMM + bias + row L2 + ReLU + BN partial sums ----------------
__global__ __launch_bounds__(256) void gemm_norm_kernel(
    const _Float16* __restrict__ xh, const _Float16* __restrict__ nh,
    const _Float16* __restrict__ wxh, const _Float16* __restrict__ wnh,
    const float* __restrict__ bx, const float* __restrict__ bn,
    float* __restrict__ out, float* __restrict__ psum, float* __restrict__ psq) {
  __shared__ _Float16 axs[16 * DD];
  __shared__ _Float16 ans[16 * DD];
  __shared__ float hbuf[16 * CC];
  __shared__ float part[16][17];
  __shared__ float invn[16];

  const int tid = threadIdx.x, rt = blockIdx.x;

  // stage A tiles (16 rows x 256 f16, both matrices)
  const uint* sx = (const uint*)(xh + (size_t)rt * 16 * DD);
  const uint* sn = (const uint*)(nh + (size_t)rt * 16 * DD);
  uint* dx = (uint*)axs; uint* dn = (uint*)ans;
  #pragma unroll
  for (int k = 0; k < 8; ++k) {
    dx[tid + k * 256] = sx[tid + k * 256];
    dn[tid + k * 256] = sn[tid + k * 256];
  }
  __syncthreads();

  const int lane = tid & 31, wv = tid >> 5;
  const int half = wv >> 2, wq = wv & 3;        // waves 0-3: x path, 4-7: neighbor path
  const _Float16* A  = half ? ans : axs;
  const _Float16* Wt = half ? wnh : wxh;
  const float* bias  = half ? bn  : bx;

  v8f acc[4] = {};
  for (int ks = 0; ks < 8; ++ks) {
    v16h a = load_a_frag(A, lane, ks);
    #pragma unroll
    for (int i = 0; i < 4; ++i) {
      int ct = wq + 4 * i;
      v16h bf = load_b_frag(Wt, ct * 16, lane, ks);
      acc[i] = __builtin_amdgcn_wmma_f32_16x16x32_f16(false, a, false, bf,
                                                      (short)0, acc[i], false, false);
    }
  }
  #pragma unroll
  for (int i = 0; i < 4; ++i) {
    int ct  = wq + 4 * i;
    int col = ct * 16 + (lane & 15);
    float bv = bias[col];
    #pragma unroll
    for (int v = 0; v < 8; ++v) {
      int row = v + ((lane >> 4) << 3);
      hbuf[row * CC + half * DOUTC + col] = acc[i][v] + bv;
    }
  }
  __syncthreads();

  // row-wise sum of squares over 512 channels
  {
    int row = tid >> 4, seg = tid & 15;
    float s = 0.f;
    #pragma unroll
    for (int k = 0; k < 32; ++k) { float v = hbuf[row * CC + seg * 32 + k]; s += v * v; }
    part[row][seg] = s;
  }
  __syncthreads();
  if (tid < 16) {
    float s = 0.f;
    #pragma unroll
    for (int k = 0; k < 16; ++k) s += part[tid][k];
    invn[tid] = 1.f / fmaxf(sqrtf(s), 1e-12f);
  }
  __syncthreads();

  // normalize + ReLU, write to d_out (coalesced), keep in LDS for channel sums
  #pragma unroll
  for (int k = 0; k < 32; ++k) {
    int idx = tid + k * 256;             // 0..8191 == row*512 + ch
    int row = idx >> 9;
    float v = fmaxf(hbuf[idx] * invn[row], 0.f);
    hbuf[idx] = v;
    out[(size_t)rt * 16 * CC + idx] = v;
  }
  __syncthreads();

  // deterministic per-tile channel partial sums for BN
  {
    int ch0 = tid * 2;
    #pragma unroll
    for (int c = 0; c < 2; ++c) {
      int ch = ch0 + c;
      float s = 0.f, ss = 0.f;
      #pragma unroll
      for (int row = 0; row < 16; ++row) { float v = hbuf[row * CC + ch]; s += v; ss += v * v; }
      psum[(size_t)rt * CC + ch] = s;
      psq [(size_t)rt * CC + ch] = ss;
    }
  }
}

// ---------------- K3: reduce partial sums -> scale/shift per channel ----------------
__global__ __launch_bounds__(256) void bn_stats_kernel(const float* __restrict__ psum,
                                                       const float* __restrict__ psq,
                                                       const float* __restrict__ gamma,
                                                       const float* __restrict__ beta,
                                                       float* __restrict__ sc,
                                                       float* __restrict__ sh) {
  __shared__ float s1[256], s2[256];
  int ch = blockIdx.x;
  float a = 0.f, b = 0.f;
  #pragma unroll
  for (int k = 0; k < 8; ++k) {
    int t = threadIdx.x + k * 256;
    a += psum[(size_t)t * CC + ch];
    b += psq [(size_t)t * CC + ch];
  }
  s1[threadIdx.x] = a; s2[threadIdx.x] = b;
  __syncthreads();
  for (int st = 128; st > 0; st >>= 1) {
    if (threadIdx.x < st) {
      s1[threadIdx.x] += s1[threadIdx.x + st];
      s2[threadIdx.x] += s2[threadIdx.x + st];
    }
    __syncthreads();
  }
  if (threadIdx.x == 0) {
    const float cnt = (float)NROWS;
    float mu  = s1[0] / cnt;
    float var = s2[0] / cnt - mu * mu;
    float scale = gamma[ch] / sqrtf(var + 1e-5f);
    sc[ch] = scale;
    sh[ch] = beta[ch] - mu * scale;
  }
}

// ---------------- K4: in-place BN affine on d_out ----------------
__global__ __launch_bounds__(256) void bn_apply_kernel(float* __restrict__ out,
                                                       const float* __restrict__ sc,
                                                       const float* __restrict__ sh) {
  size_t i = (size_t)blockIdx.x * 256 + threadIdx.x;
  int ch = (int)(i & (CC - 1));
  out[i] = out[i] * sc[ch] + sh[ch];
}

extern "C" void kernel_launch(void* const* d_in, const int* in_sizes, int n_in,
                              void* d_out, int out_size, void* d_ws, size_t ws_size,
                              hipStream_t stream) {
  const float* x     = (const float*)d_in[0];
  // d_in[1]=adj (unused), d_in[2]=p, d_in[3]=t (constants 16/8 baked in)
  const float* Wx_w  = (const float*)d_in[4];
  const float* Wx_b  = (const float*)d_in[5];
  const float* Wn_w  = (const float*)d_in[6];
  const float* Wn_b  = (const float*)d_in[7];
  const float* gamma = (const float*)d_in[8];
  const float* beta  = (const float*)d_in[9];
  float* out = (float*)d_out;

  char* ws = (char*)d_ws;
  size_t off = 0;
  _Float16* xh  = (_Float16*)(ws + off); off += (size_t)NB * NN * DD * 2;   // 16 MB
  _Float16* nh  = (_Float16*)(ws + off); off += (size_t)NB * NN * DD * 2;   // 16 MB
  _Float16* wxh = (_Float16*)(ws + off); off += (size_t)DOUTC * DD * 2;     // 128 KB
  _Float16* wnh = (_Float16*)(ws + off); off += (size_t)DOUTC * DD * 2;     // 128 KB
  float* psum   = (float*)(ws + off);    off += (size_t)NRT * CC * 4;       // 4 MB
  float* psq    = (float*)(ws + off);    off += (size_t)NRT * CC * 4;       // 4 MB
  float* sc     = (float*)(ws + off);    off += CC * 4;
  float* sh     = (float*)(ws + off);    off += CC * 4;
  (void)ws_size; (void)in_sizes; (void)n_in; (void)out_size;

  // K0: precision conversion
  cvt_x_kernel<<<(NB * NN * DD) / (256 * 4), 256, 0, stream>>>(x, xh);
  cvt_w_kernel<<<(DOUTC * DD) / 256, 256, 0, stream>>>(Wx_w, Wn_w, wxh, wnh);

  // K1: windowed KNN + neighbor mean (dynamic LDS: window f16 + sim f32 + idx)
  size_t shmem = (size_t)WW * DD * sizeof(_Float16) + 16 * WW * sizeof(float) + 32 * sizeof(int);
  knn_neib_kernel<<<dim3(MM, NB), 256, shmem, stream>>>(x, nh);

  // K2: fused dual GEMM + L2 + ReLU + BN partials
  gemm_norm_kernel<<<NRT, 256, 0, stream>>>(xh, nh, wxh, wnh, Wx_b, Wn_b, out, psum, psq);

  // K3/K4: BatchNorm
  bn_stats_kernel<<<CC, 256, 0, stream>>>(psum, psq, gamma, beta, sc, sh);
  bn_apply_kernel<<<(size_t)NROWS * CC / 256, 256, 0, stream>>>(out, sc, sh);
}
